// ImprovedUncertaintyGCN_71442486002088
// MI455X (gfx1250) — matrix operand order repaired and checked by reading, not compiled, not run on hardware
//
#include <hip/hip_runtime.h>
#include <hip/hip_bf16.h>

// ---------------- CDNA5 WMMA types ----------------
typedef __attribute__((ext_vector_type(16))) __bf16 v16bf;
typedef __attribute__((ext_vector_type(8)))  __bf16 v8bf;
typedef __attribute__((ext_vector_type(8)))  float  v8f;

static constexpr int NN     = 4096;
static constexpr int HIDC   = 512;
static constexpr int NTC    = 32;
static constexpr int HEADSC = 4;
static constexpr int DHC    = 128;

// ---------------- bf16 helpers (RNE) ----------------
__device__ __forceinline__ unsigned short f2bf_bits(float f) {
  union { float f; unsigned u; } c; c.f = f;
  unsigned u = c.u;
  u += 0x7fffu + ((u >> 16) & 1u);
  return (unsigned short)(u >> 16);
}
__device__ __forceinline__ __bf16 bits2bf(unsigned short h) {
  __bf16 r; __builtin_memcpy(&r, &h, 2); return r;
}

// ---------------- bf16 fragment loaders (contiguous k order k = 16*half + s, same for A and B), vector loads ----------------
// A 16x32 bf16: lane<16 -> row=lane, K=0..15; lane>=16 -> row=lane-16, K=16..31
__device__ __forceinline__ v16bf load_a_frag_bf(const unsigned short* __restrict__ base, int ld, int lane) {
  int half = lane >> 4, r = lane & 15;
  const unsigned short* p = base + (size_t)r * ld + half * 16;
  union { v16bf v; v8bf h[2]; } u;
  u.h[0] = *(const v8bf*)(p);
  u.h[1] = *(const v8bf*)(p + 8);
  return u.v;
}
// B 32x16 bf16 with B[k][n] = Wlike[n][k], Wlike row-major [16 x K]:
// lane<16 -> col=lane, K=0..15; lane>=16 -> col=lane-16, K=16..31 -> one contiguous 32B span.
__device__ __forceinline__ v16bf load_b_frag_bf(const unsigned short* __restrict__ base, int ld, int lane) {
  int col = lane & 15, kb = (lane >> 4) * 16;
  return *(const v16bf*)(base + (size_t)col * ld + kb);
}

// ---------------- conversion kernels ----------------
__global__ void f2bf_k(const float* __restrict__ in, unsigned short* __restrict__ out, int n) {
  int i = blockIdx.x * blockDim.x + threadIdx.x;
  if (i < n) out[i] = f2bf_bits(in[i]);
}
// transpose-convert: out[c*rows + r] = bf16(in[r*cols + c])
__global__ void f2bf_tr_k(const float* __restrict__ in, unsigned short* __restrict__ out, int rows, int cols) {
  int i = blockIdx.x * blockDim.x + threadIdx.x;
  if (i >= rows * cols) return;
  int r = i / cols, c = i - r * cols;
  out[(size_t)c * rows + r] = f2bf_bits(in[i]);
}

// ---------------- WMMA GEMM:  C[N,OUT] = A[N,K] @ W[OUT,K]^T (+bias)(+res) ----------------
template <bool BIAS, bool RES>
__global__ void wmma_gemm_bf(const unsigned short* __restrict__ A, const unsigned short* __restrict__ W,
                             const float* __restrict__ bias, const float* __restrict__ res,
                             float* __restrict__ C, int nrows, int K, int nout) {
  int wave = (int)((blockIdx.x * blockDim.x + threadIdx.x) >> 5);
  int lane = (int)(threadIdx.x & 31);
  int tnc = nout >> 4;
  int total = (nrows >> 4) * tnc;
  if (wave >= total) return;                  // wave-uniform: EXEC stays all-1s for WMMA
  int tm = wave / tnc, tn = wave - tm * tnc;
  int r0 = tm << 4, c0 = tn << 4;
  const unsigned short* Ab = A + (size_t)r0 * K;
  const unsigned short* Wb = W + (size_t)c0 * K;
  v8f acc = {};
  for (int kb = 0; kb < K; kb += 32) {
    v16bf a = load_a_frag_bf(Ab + kb, K, lane);
    v16bf b = load_b_frag_bf(Wb + kb, K, lane);
    acc = __builtin_amdgcn_wmma_f32_16x16x32_bf16(false, a, false, b, (short)0, acc, false, false);
  }
  // D layout: lane<16 rows 0..7, lane>=16 rows 8..15; col = lane&15
  int col = lane & 15, rb = (lane >> 4) << 3;
  float badd = BIAS ? bias[c0 + col] : 0.0f;
  float* out = C + (size_t)(r0 + rb) * nout + c0 + col;
  const float* rp = res + (size_t)(r0 + rb) * nout + c0 + col;  // only read if RES
#pragma unroll
  for (int v = 0; v < 8; ++v) {
    float xv = acc[v] + badd;
    if (RES) xv += rp[(size_t)v * nout];
    out[(size_t)v * nout] = xv;
  }
}

// ---------------- GCN graph kernels ----------------
__global__ void init_ones(float* __restrict__ p, int n) {
  int i = blockIdx.x * blockDim.x + threadIdx.x;
  if (i < n) p[i] = 1.0f;                       // self-loop contribution to degree
}
__global__ void deg_accum(const int* __restrict__ dst, float* __restrict__ deg, int e) {
  int i = blockIdx.x * blockDim.x + threadIdx.x;
  if (i < e) atomicAdd(&deg[dst[i]], 1.0f);
}
__global__ void to_dinv(float* __restrict__ deg, int n) {
  int i = blockIdx.x * blockDim.x + threadIdx.x;
  if (i < n) deg[i] = rsqrtf(deg[i]);           // deg >= 1 always (self loops)
}
// agg = bias + dinv[i]^2 * h[i]   (self-loop edge folded in)
__global__ void agg_init(const float* __restrict__ h, const float* __restrict__ dinv,
                         const float* __restrict__ bias, float* __restrict__ out, int total, int hid) {
  int i = blockIdx.x * blockDim.x + threadIdx.x;
  if (i >= total) return;
  int n = i / hid, f = i - n * hid;
  float d = dinv[n];
  out[i] = bias[f] + d * d * h[i];
}
// agg[dst] += dinv[src]*dinv[dst] * h[src]  (4 floats / thread; h resident in 192MB L2)
__global__ void edge_scatter(const int* __restrict__ src, const int* __restrict__ dst,
                             const float* __restrict__ dinv, const float* __restrict__ h,
                             float* __restrict__ out, int e, int hid4) {
  int idx = blockIdx.x * blockDim.x + threadIdx.x;
  if (idx >= e * hid4) return;
  int ed = idx / hid4, c = (idx - ed * hid4) * 4;
  int s = src[ed], d = dst[ed];
  float w = dinv[s] * dinv[d];
  const float* hp = h + (size_t)s * HIDC + c;
  float* op = out + (size_t)d * HIDC + c;
  atomicAdd(&op[0], w * hp[0]);
  atomicAdd(&op[1], w * hp[1]);
  atomicAdd(&op[2], w * hp[2]);
  atomicAdd(&op[3], w * hp[3]);
}
__global__ void relu_k(float* __restrict__ p, int n) {
  int i = blockIdx.x * blockDim.x + threadIdx.x;
  if (i < n) p[i] = fmaxf(p[i], 0.0f);
}
__global__ void bn_relu(float* __restrict__ z, const float* __restrict__ g, const float* __restrict__ b,
                        const float* __restrict__ mean, const float* __restrict__ var, int total, int hid) {
  int i = blockIdx.x * blockDim.x + threadIdx.x;
  if (i >= total) return;
  int f = i & (hid - 1);                        // hid = 512, power of two
  float s = g[f] * rsqrtf(var[f] + 1e-5f);
  z[i] = fmaxf((z[i] - mean[f]) * s + b[f], 0.0f);
}

// ---------------- softmax over NT=32: one wave32 per row, lane==type ----------------
__device__ __forceinline__ float wave_max32(float x) {
#pragma unroll
  for (int m = 1; m < 32; m <<= 1) x = fmaxf(x, __shfl_xor(x, m, 32));
  return x;
}
__device__ __forceinline__ float wave_sum32(float x) {
#pragma unroll
  for (int m = 1; m < 32; m <<= 1) x += __shfl_xor(x, m, 32);
  return x;
}
__global__ void softmax_uw(const float* __restrict__ logits, const float* __restrict__ tu,
                           float* __restrict__ uw, int nn) {
  int lane = threadIdx.x & 31;
  int row = (blockIdx.x * blockDim.x + threadIdx.x) >> 5;
  if (row >= nn) return;
  float x = logits[(size_t)row * NTC + lane];
  float m = wave_max32(x);
  float e = __expf(x - m);
  float s = wave_sum32(e);
  uw[(size_t)row * NTC + lane] = (e / s) * tu[lane];
}
__global__ void softmax_final(const float* __restrict__ logits, const float* __restrict__ tu,
                              float* __restrict__ p_out, float* __restrict__ unc_out, int nn) {
  int lane = threadIdx.x & 31;
  int row = (blockIdx.x * blockDim.x + threadIdx.x) >> 5;
  if (row >= nn) return;
  float x = logits[(size_t)row * NTC + lane];
  float m = wave_max32(x);
  float e = __expf(x - m);
  float s = wave_sum32(e);
  float p = e / s;
  p_out[(size_t)row * NTC + lane] = p;
  float c = wave_sum32(p * tu[lane]);
  if (lane == 0) unc_out[row] = c;
}

// ---------------- flash attention: one wave per (head, 16-query tile), 32-key blocks ----------------
// Q, K: bf16 row-major [N, 512] (heads packed).  Vt: bf16 transposed [512, N].
// __launch_bounds__(128, 1): only ~1024 waves exist in this pass (~1 wave/SIMD chip-wide),
// so give the wave the whole register file -> Q fragments live in 32 VGPRs, zero spills.
__global__ void __launch_bounds__(128, 1)
flash_attn(const unsigned short* __restrict__ Q, const unsigned short* __restrict__ Km,
           const unsigned short* __restrict__ Vt, float* __restrict__ O, int nn) {
  __shared__ __align__(16) unsigned short pbuf[4][16 * 32];  // per-wave P-tile bounce (1KB/wave)
  int w    = threadIdx.x >> 5;
  int lane = threadIdx.x & 31;
  int wave = blockIdx.x * (blockDim.x >> 5) + w;
  int qtiles = nn >> 4;
  int head = wave / qtiles;
  int qt   = wave - head * qtiles;
  int r0   = qt << 4;
  int hoff = head * DHC;
  const float scl = 0.08838834764831845f;       // 1/sqrt(128)

  const unsigned short* Kg = Km + hoff;

  // Q tile resident in registers for the whole kernel (4 x v16bf = 32 VGPRs)
  v16bf qf[4];
#pragma unroll
  for (int c = 0; c < 4; ++c)
    qf[c] = load_a_frag_bf(Q + (size_t)r0 * HIDC + hoff + c * 32, HIDC, lane);

  v16bf bones;                                   // ones B-matrix for row-sum WMMA
#pragma unroll
  for (int j = 0; j < 16; ++j) bones[j] = bits2bf(0x3F80);   // 1.0bf

  v8f ao[8];
  v8f lsum = {};
  float mi[8];
#pragma unroll
  for (int t = 0; t < 8; ++t) ao[t] = {};
#pragma unroll
  for (int v = 0; v < 8; ++v) mi[v] = -3.0e38f;

  int colc = lane & 15, mbase = (lane >> 4) << 3;
  int half = lane >> 4, r = lane & 15;

  for (int kb = 0; kb < nn; kb += 32) {
    // ---- S = Q . K^T  (two 16x16 n-tiles over this 32-key block) ----
    v8f s0 = {}, s1 = {};
#pragma unroll
    for (int c = 0; c < 4; ++c) {
      v16bf b0 = load_b_frag_bf(Kg + (size_t)(kb +  0) * HIDC + c * 32, HIDC, lane);
      v16bf b1 = load_b_frag_bf(Kg + (size_t)(kb + 16) * HIDC + c * 32, HIDC, lane);
      s0 = __builtin_amdgcn_wmma_f32_16x16x32_bf16(false, qf[c], false, b0, (short)0, s0, false, false);
      s1 = __builtin_amdgcn_wmma_f32_16x16x32_bf16(false, qf[c], false, b1, (short)0, s1, false, false);
    }
    // ---- row max: batched butterfly (8 rows per stage -> 4 DS waits/block) ----
    float a0[8], a1[8], rm[8];
#pragma unroll
    for (int v = 0; v < 8; ++v) {
      a0[v] = s0[v] * scl; a1[v] = s1[v] * scl;
      rm[v] = fmaxf(a0[v], a1[v]);
    }
#pragma unroll
    for (int st = 1; st <= 8; st <<= 1) {
      float o[8];
#pragma unroll
      for (int v = 0; v < 8; ++v) o[v] = __shfl_xor(rm[v], st, 32);
#pragma unroll
      for (int v = 0; v < 8; ++v) rm[v] = fmaxf(rm[v], o[v]);
    }
    // ---- rescale + exp + stash P to LDS ----
    float corr[8];
#pragma unroll
    for (int v = 0; v < 8; ++v) {
      float m2 = fmaxf(mi[v], rm[v]);
      corr[v] = __expf(mi[v] - m2);
      mi[v] = m2;
    }
#pragma unroll
    for (int v = 0; v < 8; ++v) {
      float p0 = __expf(a0[v] - mi[v]);
      float p1 = __expf(a1[v] - mi[v]);
      int row = mbase + v;
      pbuf[w][row * 32 + colc]      = f2bf_bits(p0);
      pbuf[w][row * 32 + 16 + colc] = f2bf_bits(p1);
    }
#pragma unroll
    for (int v = 0; v < 8; ++v) {
      lsum[v] *= corr[v];
#pragma unroll
      for (int t = 0; t < 8; ++t) ao[t][v] *= corr[v];
    }
    // ---- rebuild P as an A-fragment via two 16B DS loads ----
    const unsigned short* pb = &pbuf[w][r * 32 + half * 16];
    union { v16bf v; v8bf h[2]; } up;
    up.h[0] = *(const v8bf*)(pb);
    up.h[1] = *(const v8bf*)(pb + 8);
    v16bf pa = up.v;
    // ---- row-sum via WMMA against ones; AO += P.V ----
    lsum = __builtin_amdgcn_wmma_f32_16x16x32_bf16(false, pa, false, bones, (short)0, lsum, false, false);
#pragma unroll
    for (int t = 0; t < 8; ++t) {
      v16bf bv = load_b_frag_bf(Vt + (size_t)(hoff + t * 16) * nn + kb, nn, lane);
      ao[t] = __builtin_amdgcn_wmma_f32_16x16x32_bf16(false, pa, false, bv, (short)0, ao[t], false, false);
    }
  }
  float inv[8];
#pragma unroll
  for (int v = 0; v < 8; ++v) inv[v] = 1.0f / lsum[v];       // row sum replicated across lanes
#pragma unroll
  for (int t = 0; t < 8; ++t)
#pragma unroll
    for (int v = 0; v < 8; ++v)
      O[(size_t)(r0 + mbase + v) * HIDC + hoff + t * 16 + colc] = ao[t][v] * inv[v];
}

// ---------------- host orchestration ----------------
extern "C" void kernel_launch(void* const* d_in, const int* in_sizes, int n_in,
                              void* d_out, int out_size, void* d_ws, size_t ws_size,
                              hipStream_t stream) {
  const float* x    = (const float*)d_in[0];
  const int*   ei   = (const int*)d_in[1];
  const float* W1   = (const float*)d_in[2];  const float* b1  = (const float*)d_in[3];
  const float* W2   = (const float*)d_in[4];  const float* b2  = (const float*)d_in[5];
  const float* W3   = (const float*)d_in[6];  const float* b3  = (const float*)d_in[7];
  const float* mw1  = (const float*)d_in[8];  const float* mb1 = (const float*)d_in[9];
  const float* bng  = (const float*)d_in[10]; const float* bnb = (const float*)d_in[11];
  const float* bnm  = (const float*)d_in[12]; const float* bnv = (const float*)d_in[13];
  const float* mw2  = (const float*)d_in[14]; const float* mb2 = (const float*)d_in[15];
  const float* tu   = (const float*)d_in[16];
  const float* uew  = (const float*)d_in[17]; const float* ueb = (const float*)d_in[18];
  const float* wqkv = (const float*)d_in[19]; const float* bqkv= (const float*)d_in[20];
  const float* wo   = (const float*)d_in[21]; const float* bo  = (const float*)d_in[22];
  const int E = in_sizes[1] / 2;
  const int* srcp = ei;
  const int* dstp = ei + E;

  // ---- workspace layout ----
  const size_t NB = (size_t)NN * HIDC;          // 2M elements
  float* base = (float*)d_ws;
  float* B0 = base;       float* B1 = B0 + NB; float* B2 = B1 + NB;
  float* B3 = B2 + NB;    float* B4 = B3 + NB; float* B5 = B4 + NB;
  float* dinv = B5 + NB;
  float* L0   = dinv + NN;                      // logits [N,32]
  float* UW   = L0 + (size_t)NN * NTC;          // uncertainty-weighted p [N,32]
  unsigned short* Abf = (unsigned short*)(UW + (size_t)NN * NTC);  // bf16 A staging [N,512]
  unsigned short* Wbf = Abf + NB;                                  // bf16 W staging [512,512]
  unsigned short* Qbf = Wbf + (size_t)HIDC * HIDC;                 // bf16 Q [N,512]
  unsigned short* Kbf = Qbf + NB;                                  // bf16 K [N,512]
  unsigned short* Vtb = Kbf + NB;                                  // bf16 V^T [512,N]

  auto cd = [](int a, int b) { return (a + b - 1) / b; };
  auto gemm = [&](const float* A, const float* W, const float* bias, const float* res,
                  float* C, int n, int K, int out) {
    f2bf_k<<<cd(n * K, 256), 256, 0, stream>>>(A, Abf, n * K);
    f2bf_k<<<cd(out * K, 256), 256, 0, stream>>>(W, Wbf, out * K);
    int blocks = cd((n / 16) * (out / 16), 4);
    if (bias && res)
      wmma_gemm_bf<true, true><<<blocks, 128, 0, stream>>>(Abf, Wbf, bias, res, C, n, K, out);
    else if (bias)
      wmma_gemm_bf<true, false><<<blocks, 128, 0, stream>>>(Abf, Wbf, bias, nullptr, C, n, K, out);
    else
      wmma_gemm_bf<false, false><<<blocks, 128, 0, stream>>>(Abf, Wbf, nullptr, nullptr, C, n, K, out);
  };

  // ---- GCN normalization: deg -> dinv ----
  init_ones<<<cd(NN, 256), 256, 0, stream>>>(dinv, NN);
  deg_accum<<<cd(E, 256), 256, 0, stream>>>(dstp, dinv, E);
  to_dinv<<<cd(NN, 256), 256, 0, stream>>>(dinv, NN);

  const int TOT = NN * HIDC;
  const int ETH = E * (HIDC / 4);

  // ---- layer 1: h = relu(agg(x @ W1^T)) ----
  gemm(x, W1, nullptr, nullptr, B0, NN, HIDC, HIDC);
  agg_init<<<cd(TOT, 256), 256, 0, stream>>>(B0, dinv, b1, B1, TOT, HIDC);
  edge_scatter<<<cd(ETH, 256), 256, 0, stream>>>(srcp, dstp, dinv, B0, B1, E, HIDC / 4);
  relu_k<<<cd(TOT, 256), 256, 0, stream>>>(B1, TOT);
  // ---- layer 2 ----
  gemm(B1, W2, nullptr, nullptr, B0, NN, HIDC, HIDC);
  agg_init<<<cd(TOT, 256), 256, 0, stream>>>(B0, dinv, b2, B1, TOT, HIDC);
  edge_scatter<<<cd(ETH, 256), 256, 0, stream>>>(srcp, dstp, dinv, B0, B1, E, HIDC / 4);
  relu_k<<<cd(TOT, 256), 256, 0, stream>>>(B1, TOT);
  // ---- layer 3 -> h3 in B2 ----
  gemm(B1, W3, nullptr, nullptr, B0, NN, HIDC, HIDC);
  agg_init<<<cd(TOT, 256), 256, 0, stream>>>(B0, dinv, b3, B2, TOT, HIDC);
  edge_scatter<<<cd(ETH, 256), 256, 0, stream>>>(srcp, dstp, dinv, B0, B2, E, HIDC / 4);
  relu_k<<<cd(TOT, 256), 256, 0, stream>>>(B2, TOT);

  // ---- MLP #1 -> logits_initial -> uw ----
  gemm(B2, mw1, mb1, nullptr, B0, NN, HIDC, HIDC);
  bn_relu<<<cd(TOT, 256), 256, 0, stream>>>(B0, bng, bnb, bnm, bnv, TOT, HIDC);
  gemm(B0, mw2, mb2, nullptr, L0, NN, HIDC, NTC);
  softmax_uw<<<cd(NN * 32, 128), 128, 0, stream>>>(L0, tu, UW, NN);

  // ---- us = uw @ ue_w^T + ue_b ; q_in = h3 + us (residual fused) -> B1 ----
  gemm(UW, uew, ueb, B2, B1, NN, NTC, HIDC);

  // ---- QKV projections ----
  gemm(B1, wqkv,                           bqkv,            nullptr, B3, NN, HIDC, HIDC); // Q from q_in
  gemm(B2, wqkv + (size_t)HIDC * HIDC,     bqkv + HIDC,     nullptr, B4, NN, HIDC, HIDC); // K from h3
  gemm(B2, wqkv + (size_t)2 * HIDC * HIDC, bqkv + 2 * HIDC, nullptr, B5, NN, HIDC, HIDC); // V from h3

  // ---- bf16 staging for attention: Q,K row-major; V transposed ----
  f2bf_k<<<cd(TOT, 256), 256, 0, stream>>>(B3, Qbf, TOT);
  f2bf_k<<<cd(TOT, 256), 256, 0, stream>>>(B4, Kbf, TOT);
  f2bf_tr_k<<<cd(TOT, 256), 256, 0, stream>>>(B5, Vtb, NN, HIDC);

  // ---- flash attention: 4 heads x 256 query tiles = 1024 waves ----
  flash_attn<<<(HEADSC * (NN / 16)) / 4, 128, 0, stream>>>(Qbf, Kbf, Vtb, B0, NN);

  // ---- out-proj + residual: h_final = h3 + ao @ wo^T + bo -> B1 ----
  gemm(B0, wo, bo, B2, B1, NN, HIDC, HIDC);

  // ---- MLP #2 -> final logits -> p, total_uncertainty ----
  gemm(B1, mw1, mb1, nullptr, B0, NN, HIDC, HIDC);
  bn_relu<<<cd(TOT, 256), 256, 0, stream>>>(B0, bng, bnb, bnm, bnv, TOT, HIDC);
  gemm(B0, mw2, mb2, nullptr, L0, NN, HIDC, NTC);

  float* p_out   = (float*)d_out;
  float* unc_out = p_out + (size_t)NN * NTC;
  softmax_final<<<cd(NN * 32, 128), 128, 0, stream>>>(L0, tu, p_out, unc_out, NN);
}
